// SwinTransformerBlock_84868553769506
// MI455X (gfx1250) — compile-verified, matrix operand order checked by reading, not run on hardware
//
#include <hip/hip_runtime.h>
#include <math.h>

// ---------------------------------------------------------------------------
// Swin Transformer block on MI455X (gfx1250, wave32, WMMA + TDM).
// GEMMs: v_wmma_f32_16x16x32_f16 (f16 in, f32 acc). B (weight) tiles staged
// to LDS with TENSOR_LOAD_TO_LDS (Tensor Data Mover, TENSORcnt) for 4x reuse
// across the block's M-waves; A fragments stream from global (hot in 192MB
// L2) with global_prefetch_b8.  Intermediates in f16: block is BW-bound
// (~168 GFLOP vs >200MB fp32 activations @ 23.3 TB/s HBM).
// ---------------------------------------------------------------------------

typedef __attribute__((ext_vector_type(16))) _Float16 v16h;
typedef __attribute__((ext_vector_type(8)))  _Float16 v8h;
typedef __attribute__((ext_vector_type(8)))  float    v8f;
typedef __attribute__((ext_vector_type(4)))  unsigned int u32x4;
typedef __attribute__((ext_vector_type(8)))  int      i32x8;
typedef __attribute__((ext_vector_type(4)))  int      i32x4;

#define BQ   32
#define HQ   56
#define WQ   56
#define CQ   256
#define WS7  7
#define NTK  49
#define NWIN 2048
#define NH   8
#define HD   32
#define MROWS 100352     // B*H*W == NWIN*49, divisible by 128
#define MLPD 1024
#define SCALEQ 0.17677669529663687f   // 1/sqrt(32)

// --- TDM availability / arity detection (per CDNA5_HIP.md probe notes) -----
#if defined(__has_builtin)
#if __has_builtin(__builtin_amdgcn_tensor_load_to_lds) && \
    __has_builtin(__builtin_amdgcn_s_wait_tensorcnt)
#define USE_TDM 1
#endif
#endif
#ifndef USE_TDM
#define USE_TDM 0
#endif
#if __has_include(<hip/amd_detail/amd_gfx1250_TDM.h>)
#define TDM_ARITY6 1   // therock/clang-23 headers present -> 6-arg builtin
#else
#define TDM_ARITY6 0   // ROCm 7.2 clang-22 -> 5-arg builtin
#endif

// ---------------------------------------------------------------------------
// Fragment loader. CDNA5 16-bit A layout (ISA 7.12.2): lane (L&15) = row,
// K split {0..7,16..23} / {8..15,24..31} by lane<16 => two 16B loads.
// B fragments mirror it (lane = column, contiguous K), so weights are stored
// transposed [N,K] and the same loader serves both (global or LDS pointer).
// ---------------------------------------------------------------------------
__device__ __forceinline__ v16h load_frag(const _Float16* __restrict__ row,
                                          int kb, int lane) {
  const int s = (lane & 16) ? 8 : 0;
  v8h lo = *(const v8h*)(row + kb + s);
  v8h hi = *(const v8h*)(row + kb + 16 + s);
  v16h r;
#pragma unroll
  for (int i = 0; i < 8; ++i) { r[i] = lo[i]; r[i + 8] = hi[i]; }
  return r;
}

__device__ __forceinline__ v16h zero_frag() {
  v16h z;
#pragma unroll
  for (int i = 0; i < 16; ++i) z[i] = (_Float16)0.f;
  return z;
}

// windowed row -> original token row (window-reverse + un-shift, roll +3)
__device__ __forceinline__ int remap_row(int r) {
  int win = r / NTK, t = r - win * NTK;
  int b = win >> 6, wrem = win & 63;
  int wi = wrem >> 3, wj = wrem & 7;
  int ti = t / WS7, tj = t - ti * WS7;
  int h = wi * WS7 + ti + 3; if (h >= HQ) h -= HQ;
  int w = wj * WS7 + tj + 3; if (w >= WQ) w -= WQ;
  return b * (HQ * WQ) + h * WQ + w;
}

#if USE_TDM
// ---------------------------------------------------------------------------
// TDM descriptor (ISA 08_async_tensor §8): copy a [64 x 256] f16 tile,
// row-major with row stride K elements, from global into LDS at lds_off.
// group0: count=1 | lds_addr | global tile addr | type=2 ("image")
// group1: data_size=2B | tensor_dim0=K | tensor_dim1=big | tile 256x64 |
//         tensor_dim0_stride=K | tensor_dim1_stride=K
// ---------------------------------------------------------------------------
__device__ __forceinline__ void tdm_load_b_tile(unsigned int lds_off,
                                                const _Float16* gsrc, int K) {
  const unsigned long long ga = (unsigned long long)(size_t)gsrc;
  u32x4 g0;
  g0[0] = 1u;                                            // count=1, user D#
  g0[1] = lds_off;                                       // lds_addr (bytes)
  g0[2] = (unsigned int)(ga & 0xFFFFFFFFu);              // global_addr lo
  g0[3] = (unsigned int)((ga >> 32) & 0x1FFFFFFu) | (2u << 30); // hi | type=2
  const unsigned int dim1 = 1u << 30;                    // large (no OOB)
  i32x8 g1;
  g1[0] = (int)(1u << 16);                               // data_size=2B
  g1[1] = (int)((K & 0xFFFF) << 16);                     // tensor_dim0 lo16
  g1[2] = (int)(((K >> 16) & 0xFFFF) | ((dim1 & 0xFFFF) << 16));
  g1[3] = (int)(((dim1 >> 16) & 0xFFFF) | (256u << 16)); // tile_dim0 = 256
  g1[4] = 64;                                            // tile_dim1 = 64
  g1[5] = K;                                             // dim0_stride lo32
  g1[6] = (int)((K & 0xFFFF) << 16);                     // dim1_stride lo16
  g1[7] = (int)((unsigned)K >> 16);                      // dim1_stride hi
  i32x4 gz = {0, 0, 0, 0};
#if TDM_ARITY6
  i32x8 gz8 = {0, 0, 0, 0, 0, 0, 0, 0};
  __builtin_amdgcn_tensor_load_to_lds(g0, g1, gz, gz, gz8, 0);
#else
  __builtin_amdgcn_tensor_load_to_lds(g0, g1, gz, gz, 0);
#endif
}
#endif

// ---------------------------------------------------------------------------
// Weight convert + transpose:  wt[n*K + k] = (f16) w[k*N + n]
// ---------------------------------------------------------------------------
__global__ void cvt_transpose(const float* __restrict__ w,
                              _Float16* __restrict__ wt, int K, int N) {
  int i = blockIdx.x * 256 + threadIdx.x;
  if (i < K * N) {
    int n = i / K, k = i - n * K;
    wt[i] = (_Float16)w[(size_t)k * N + n];
  }
}

// ---------------------------------------------------------------------------
// LayerNorm (one wave per token, 8 channels/lane) with optional fused
// cyclic-shift + window-partition source remap (ln1 path).
// ---------------------------------------------------------------------------
__global__ __launch_bounds__(256) void ln_kernel(
    const float* __restrict__ x, const float* __restrict__ gam,
    const float* __restrict__ bet, _Float16* __restrict__ out, int windowed) {
  const int lane = threadIdx.x & 31;
  const int row  = blockIdx.x * 8 + (threadIdx.x >> 5);
  int srow = row;
  if (windowed) {
    int win = row / NTK, t = row - win * NTK;
    int b = win >> 6, wrem = win & 63;
    int wi = wrem >> 3, wj = wrem & 7;
    int ti = t / WS7, tj = t - ti * WS7;
    int hs = wi * WS7 + ti + 3; if (hs >= HQ) hs -= HQ;   // roll(-3) read
    int ws = wj * WS7 + tj + 3; if (ws >= WQ) ws -= WQ;
    srow = b * (HQ * WQ) + hs * WQ + ws;
  }
  const float* xr = x + (size_t)srow * CQ;
  float v[8];
#pragma unroll
  for (int i = 0; i < 8; ++i) v[i] = xr[lane + 32 * i];
  float s = 0.f;
#pragma unroll
  for (int i = 0; i < 8; ++i) s += v[i];
#pragma unroll
  for (int m = 16; m; m >>= 1) s += __shfl_xor(s, m, 32);
  const float mean = s * (1.f / 256.f);
  float q = 0.f;
#pragma unroll
  for (int i = 0; i < 8; ++i) { float d = v[i] - mean; q += d * d; }
#pragma unroll
  for (int m = 16; m; m >>= 1) q += __shfl_xor(q, m, 32);
  const float inv = rsqrtf(q * (1.f / 256.f) + 1e-5f);
  _Float16* orow = out + (size_t)row * CQ;
#pragma unroll
  for (int i = 0; i < 8; ++i) {
    int c = lane + 32 * i;
    orow[c] = (_Float16)((v[i] - mean) * inv * gam[c] + bet[c]);
  }
}

// ---------------------------------------------------------------------------
// WMMA GEMM:  C[M,N] = A[M,K] * Wt[N,K]^T + bias.  256 thr = 8 waves 4Mx2N;
// block tile 128x64, wave tile 32x32 (2x2 WMMA).  B tile staged to LDS by
// the Tensor Data Mover in 256-wide K chunks; A streamed from global.
// Epilogues: 0 f16 (+bias) | 1 proj remap+residual f32 | 2 GELU f16 |
//            3 +residual f32.
// ---------------------------------------------------------------------------
template <int EPI>
__global__ __launch_bounds__(256) void gemm_wmma(
    const _Float16* __restrict__ A, const _Float16* __restrict__ Wt,
    const float* __restrict__ bias, void* __restrict__ out,
    const float* __restrict__ aux, int N, int K) {
  __shared__ __align__(16) _Float16 Bs[64][256];   // 32 KB B tile

  const int lane = threadIdx.x & 31;
  const int wave = threadIdx.x >> 5;
  const int mbase = blockIdx.x * 128 + (wave >> 1) * 32;
  const int nblk  = blockIdx.y * 64;
  const int nloc  = (wave & 1) * 32;

  const _Float16* a0 = A  + (size_t)(mbase + (lane & 15)) * K;
  const _Float16* a1 = a0 + (size_t)16 * K;
  const _Float16* br0 = &Bs[nloc + (lane & 15)][0];
  const _Float16* br1 = &Bs[nloc + 16 + (lane & 15)][0];

  v8f acc[2][2];
#pragma unroll
  for (int i = 0; i < 2; ++i)
#pragma unroll
    for (int j = 0; j < 2; ++j) acc[i][j] = {};

  for (int kc = 0; kc < K; kc += 256) {            // K is 256 or 1024
#if USE_TDM
    if (wave == 0) {                               // one DMA per block/chunk
      tdm_load_b_tile((unsigned int)(size_t)&Bs[0][0],
                      Wt + (size_t)nblk * K + kc, K);
      __builtin_amdgcn_s_wait_tensorcnt(0);
    }
#else
    for (int u = threadIdx.x; u < 64 * 32; u += 256) {  // 16B units
      int rrow = u >> 5, rcol = (u & 31) * 8;
      *(v8h*)&Bs[rrow][rcol] =
          *(const v8h*)(Wt + (size_t)(nblk + rrow) * K + kc + rcol);
    }
#endif
    __syncthreads();
#pragma unroll 4
    for (int kb = 0; kb < 256; kb += 32) {
      if (kc + kb + 32 < K) {                      // gfx1250 global_prefetch
        __builtin_prefetch(a0 + kc + kb + 32, 0, 1);
        __builtin_prefetch(a1 + kc + kb + 32, 0, 1);
      }
      v16h fa0 = load_frag(a0, kc + kb, lane);
      v16h fa1 = load_frag(a1, kc + kb, lane);
      v16h fb0 = load_frag(br0, kb, lane);         // ds_load_b128 x2
      v16h fb1 = load_frag(br1, kb, lane);
      acc[0][0] = __builtin_amdgcn_wmma_f32_16x16x32_f16(false, fa0, false, fb0,
                                                         (short)0, acc[0][0], false, false);
      acc[0][1] = __builtin_amdgcn_wmma_f32_16x16x32_f16(false, fa0, false, fb1,
                                                         (short)0, acc[0][1], false, false);
      acc[1][0] = __builtin_amdgcn_wmma_f32_16x16x32_f16(false, fa1, false, fb0,
                                                         (short)0, acc[1][0], false, false);
      acc[1][1] = __builtin_amdgcn_wmma_f32_16x16x32_f16(false, fa1, false, fb1,
                                                         (short)0, acc[1][1], false, false);
    }
    __syncthreads();                               // before next chunk's DMA
  }

  const int colOff = lane & 15;
  const int rowOff = (lane & 16) ? 8 : 0;          // lanes 16-31 hold M+8
#pragma unroll
  for (int i = 0; i < 2; ++i)
#pragma unroll
    for (int j = 0; j < 2; ++j) {
      const int col = nblk + nloc + j * 16 + colOff;
      const float bv = bias[col];
#pragma unroll
      for (int r = 0; r < 8; ++r) {
        const int row = mbase + i * 16 + rowOff + r;
        float v = acc[i][j][r] + bv;
        if (EPI == 0) {
          ((_Float16*)out)[(size_t)row * N + col] = (_Float16)v;
        } else if (EPI == 1) {
          int orow = remap_row(row);
          ((float*)out)[(size_t)orow * N + col] = v + aux[(size_t)orow * N + col];
        } else if (EPI == 2) {
          float g = 0.5f * v * (1.0f + erff(v * 0.70710678118654752f));
          ((_Float16*)out)[(size_t)row * N + col] = (_Float16)g;
        } else {
          ((float*)out)[(size_t)row * N + col] = v + aux[(size_t)row * N + col];
        }
      }
    }
}

// ---------------------------------------------------------------------------
// Windowed attention: one workgroup (128 thr = 4 waves) per (window, head).
// 49 tokens padded to 64.  HD==32 => S tile is a single K=32 WMMA.
// Softmax + relative-position bias via LDS; padded keys masked with -3e4.
// ---------------------------------------------------------------------------
__global__ __launch_bounds__(128) void attn_kernel(
    const _Float16* __restrict__ qkv,   // [MROWS, 768] windowed order
    const float* __restrict__ rpb,      // [169, 8]
    _Float16* __restrict__ o16) {       // [MROWS, 256]
  __shared__ __align__(16) float    Sf[64][64];
  __shared__ __align__(16) _Float16 Pm[64][64];
  __shared__ __align__(16) _Float16 Vt[32][64];    // V transposed [dim][token]

  const int head = blockIdx.x & 7;
  const int win  = blockIdx.x >> 3;
  const int tid  = threadIdx.x;
  const int lane = tid & 31;
  const int wave = tid >> 5;
  const size_t base = (size_t)win * NTK * 768;

  for (int idx = tid; idx < 64 * 32; idx += 128) {
    int t = idx >> 5, d = idx & 31;
    _Float16 val = (_Float16)0.f;
    if (t < NTK) val = qkv[base + (size_t)t * 768 + 512 + head * HD + d];
    Vt[d][t] = val;
  }

  const int rq = 16 * wave + (lane & 15);
  v16h aq = (rq < NTK)
                ? load_frag(qkv + base + (size_t)rq * 768 + head * HD, 0, lane)
                : zero_frag();
#pragma unroll
  for (int nt = 0; nt < 4; ++nt) {
    const int cn = nt * 16 + (lane & 15);
    v16h bk = (cn < NTK)
                  ? load_frag(qkv + base + (size_t)cn * 768 + 256 + head * HD, 0, lane)
                  : zero_frag();
    v8f s = {};
    s = __builtin_amdgcn_wmma_f32_16x16x32_f16(false, aq, false, bk, (short)0,
                                               s, false, false);
    const int col = nt * 16 + (lane & 15);
    const int rb  = (lane & 16) ? 8 : 0;
#pragma unroll
    for (int r = 0; r < 8; ++r) Sf[16 * wave + rb + r][col] = s[r];
  }
  __syncthreads();

  if (tid < 64) {
    const int t  = tid;
    const int tq = (t < NTK) ? t : 0;
    const int qi = tq / WS7, qj = tq - qi * WS7;
    float sv[64];
    float mx = -3.0e38f;
    for (int m = 0; m < 64; ++m) {
      float s;
      if (m < NTK) {
        int ki = m / WS7, kj = m - ki * WS7;
        int idx = (qi - ki + 6) * 13 + (qj - kj + 6);
        s = Sf[t][m] * SCALEQ + rpb[idx * NH + head];
      } else {
        s = -3.0e4f;
      }
      sv[m] = s;
      mx = fmaxf(mx, s);
    }
    float sum = 0.f;
    for (int m = 0; m < 64; ++m) { float e = __expf(sv[m] - mx); sv[m] = e; sum += e; }
    const float inv = 1.f / sum;
    for (int m = 0; m < 64; ++m) Pm[t][m] = (_Float16)(sv[m] * inv);
  }
  __syncthreads();

  v8f o0 = {}, o1 = {};
  const int rp = 16 * wave + (lane & 15);
#pragma unroll
  for (int kb = 0; kb < 64; kb += 32) {
    v16h ap  = load_frag(&Pm[rp][0], kb, lane);
    v16h fb0 = load_frag(&Vt[lane & 15][0], kb, lane);
    v16h fb1 = load_frag(&Vt[16 + (lane & 15)][0], kb, lane);
    o0 = __builtin_amdgcn_wmma_f32_16x16x32_f16(false, ap, false, fb0, (short)0,
                                                o0, false, false);
    o1 = __builtin_amdgcn_wmma_f32_16x16x32_f16(false, ap, false, fb1, (short)0,
                                                o1, false, false);
  }
  const int rb = (lane & 16) ? 8 : 0;
#pragma unroll
  for (int r = 0; r < 8; ++r) {
    const int t = 16 * wave + rb + r;
    if (t < NTK) {
      const size_t orow = (size_t)(win * NTK + t) * CQ + head * HD;
      o16[orow + (lane & 15)]      = (_Float16)o0[r];
      o16[orow + 16 + (lane & 15)] = (_Float16)o1[r];
    }
  }
}

// ---------------------------------------------------------------------------
// Host-side orchestration (all on `stream`, graph-capture safe).
// ---------------------------------------------------------------------------
extern "C" void kernel_launch(void* const* d_in, const int* in_sizes, int n_in,
                              void* d_out, int out_size, void* d_ws,
                              size_t ws_size, hipStream_t stream) {
  (void)in_sizes; (void)n_in; (void)out_size; (void)ws_size;
  const float* x      = (const float*)d_in[0];
  const float* qkv_w  = (const float*)d_in[1];
  const float* qkv_b  = (const float*)d_in[2];
  const float* proj_w = (const float*)d_in[3];
  const float* proj_b = (const float*)d_in[4];
  const float* rpb    = (const float*)d_in[5];
  const float* n1g    = (const float*)d_in[6];
  const float* n1b    = (const float*)d_in[7];
  const float* n2g    = (const float*)d_in[8];
  const float* n2b    = (const float*)d_in[9];
  const float* w1     = (const float*)d_in[10];
  const float* b1     = (const float*)d_in[11];
  const float* w2     = (const float*)d_in[12];
  const float* b2     = (const float*)d_in[13];
  float* out = (float*)d_out;

  char* ws = (char*)d_ws;
  const size_t SZ_Y   = (size_t)MROWS * CQ * 2;
  const size_t SZ_QKV = (size_t)MROWS * 768 * 2;
  const size_t SZ_O   = (size_t)MROWS * CQ * 2;
  const size_t SZ_X2  = (size_t)MROWS * CQ * 4;
  _Float16* y16   = (_Float16*)(ws);                    // also h16 (ln2 out)
  _Float16* qkv16 = (_Float16*)(ws + SZ_Y);
  _Float16* o16   = (_Float16*)(ws + SZ_Y + SZ_QKV);
  float*    x2    = (float*)   (ws + SZ_Y + SZ_QKV + SZ_O);
  char*     wbase =            ws + SZ_Y + SZ_QKV + SZ_O + SZ_X2;
  _Float16* qkv_wt = (_Float16*)(wbase);                          // [768,256]
  _Float16* proj_wt= (_Float16*)(wbase + 768 * 256 * 2);          // [256,256]
  _Float16* w1t    = (_Float16*)(wbase + (768 + 256) * 256 * 2);  // [1024,256]
  _Float16* w2t    = (_Float16*)(wbase + (768 + 256 + 1024) * 256 * 2); // [256,1024]
  _Float16* g16 = qkv16;   // MLP hidden reuses dead qkv16+o16 region

  cvt_transpose<<<(256 * 768 + 255) / 256, 256, 0, stream>>>(qkv_w, qkv_wt, CQ, 768);
  cvt_transpose<<<(256 * 256 + 255) / 256, 256, 0, stream>>>(proj_w, proj_wt, CQ, CQ);
  cvt_transpose<<<(256 * 1024 + 255) / 256, 256, 0, stream>>>(w1, w1t, CQ, MLPD);
  cvt_transpose<<<(1024 * 256 + 255) / 256, 256, 0, stream>>>(w2, w2t, MLPD, CQ);

  ln_kernel<<<MROWS / 8, 256, 0, stream>>>(x, n1g, n1b, y16, 1);

  gemm_wmma<0><<<dim3(MROWS / 128, 768 / 64), 256, 0, stream>>>(
      y16, qkv_wt, qkv_b, qkv16, nullptr, 768, CQ);

  attn_kernel<<<NWIN * NH, 128, 0, stream>>>(qkv16, rpb, o16);

  gemm_wmma<1><<<dim3(MROWS / 128, CQ / 64), 256, 0, stream>>>(
      o16, proj_wt, proj_b, x2, x, CQ, CQ);

  ln_kernel<<<MROWS / 8, 256, 0, stream>>>(x2, n2g, n2b, y16, 0);

  gemm_wmma<2><<<dim3(MROWS / 128, MLPD / 64), 256, 0, stream>>>(
      y16, w1t, b1, g16, nullptr, MLPD, CQ);

  gemm_wmma<3><<<dim3(MROWS / 128, CQ / 64), 256, 0, stream>>>(
      g16, w2t, b2, out, x2, CQ, MLPD);
}
